// Scheduler_87505663688923
// MI455X (gfx1250) — matrix operand order repaired and checked by reading, not compiled, not run on hardware
//
#include <hip/hip_runtime.h>
#include <cstddef>
#include <cstdint>

// ---------------------------------------------------------------------------
// Problem constants (match reference)
// ---------------------------------------------------------------------------
#define BB    64
#define NCN   8
#define NPN   256
#define EMB   256
#define HH    8
#define DD    32
#define EPT   65536        // edges per type (B * 1024)
#define NCT   512          // B*NC crane nodes
#define NPT   16384        // B*NP pile nodes
#define TOT   16896        // NCT + NPT
#define NROWS 131072       // B*NP*NC actor rows

// ---------------------------------------------------------------------------
// WMMA types (gfx1250, wave32)
// ---------------------------------------------------------------------------
typedef __attribute__((ext_vector_type(16))) __bf16        v16bf;
typedef __attribute__((ext_vector_type(8)))  float         v8f;
typedef __attribute__((ext_vector_type(8)))  unsigned int  v8u;

union BFPack { v8u u; v16bf b; };

__device__ __forceinline__ unsigned short f2bf(float f) {
    unsigned int u = __float_as_uint(f);
    unsigned int r = (u + 0x7fffu + ((u >> 16) & 1u)) >> 16;   // RNE
    return (unsigned short)r;
}
__device__ __forceinline__ unsigned int pk2bf(float lo, float hi) {
    return ((unsigned int)f2bf(hi) << 16) | (unsigned int)f2bf(lo);
}
__device__ __forceinline__ float elu1(float x) {
    return x > 0.f ? x : __expf(x) - 1.f;
}

// ---------------------------------------------------------------------------
// Generic WMMA GEMM:  C[M,N] = act( gate*(A[M,K]@W[K,N] + bias) + (1-g)*resid )
// Block tile 128(M) x 64(N), BK=32, 256 threads (8 waves); wave -> 16x64.
// float4 global loads -> bf16 pack -> ds_store_b64/b32; double-buffered LDS.
// Launch guarantees: M,N arbitrary (guarded); K guarded (tail tile only).
// act: 0=none, 1=elu.  resid!=null enables gated skip (gate scalar ptr).
// ---------------------------------------------------------------------------
__global__ __launch_bounds__(256) void k_gemm(
    const float* __restrict__ A, const float* __restrict__ W,
    const float* __restrict__ bias, const float* __restrict__ resid,
    const float* __restrict__ gate, float* __restrict__ C,
    int M, int K, int N, int act)
{
    __shared__ __align__(16) unsigned short As[2][128][36];
    __shared__ __align__(16) unsigned short Bs[2][64][36];   // Bs[n][k]
    const int tid  = threadIdx.x;
    const int wave = tid >> 5, lane = tid & 31;
    const int half = lane >> 4, l16 = lane & 15;
    const int m0 = blockIdx.x * 128;
    const int n0 = blockIdx.y * 64;

    auto stageA = [&](int bf_, int kk) {
        for (int i = tid; i < 128 * 8; i += 256) {
            int r = i >> 3, c4 = (i & 7) << 2;
            int gm = m0 + r, gk = kk + c4;
            float4 v = make_float4(0.f, 0.f, 0.f, 0.f);
            if (gm < M) {
                const float* ap = A + (size_t)gm * K + gk;
                if (gk + 4 <= K) {
                    v = *(const float4*)ap;            // global_load_b128
                } else {
                    if (gk     < K) v.x = ap[0];
                    if (gk + 1 < K) v.y = ap[1];
                    if (gk + 2 < K) v.z = ap[2];
                    if (gk + 3 < K) v.w = ap[3];
                }
            }
            uint2 p; p.x = pk2bf(v.x, v.y); p.y = pk2bf(v.z, v.w);
            *(uint2*)&As[bf_][r][c4] = p;              // ds_store_b64
        }
    };
    auto stageB = [&](int bf_, int kk) {
        // unit: 4 n-values x 2 k-values; 16 ngroups * 16 kpairs = 256 units
        for (int i = tid; i < 256; i += 256) {
            int ng = i & 15, kp = i >> 4;
            int n = ng << 2, k2 = kp << 1;
            int gk = kk + k2;
            float4 r0 = make_float4(0.f, 0.f, 0.f, 0.f), r1 = r0;
            const float* w0 = W + (size_t)gk * N + n0 + n;
            if (gk     < K) r0 = *(const float4*)w0;
            if (gk + 1 < K) r1 = *(const float4*)(w0 + N);
            *(unsigned int*)&Bs[bf_][n + 0][k2] = pk2bf(r0.x, r1.x);
            *(unsigned int*)&Bs[bf_][n + 1][k2] = pk2bf(r0.y, r1.y);
            *(unsigned int*)&Bs[bf_][n + 2][k2] = pk2bf(r0.z, r1.z);
            *(unsigned int*)&Bs[bf_][n + 3][k2] = pk2bf(r0.w, r1.w);
        }
    };

    v8f acc[4];
    #pragma unroll
    for (int s = 0; s < 4; ++s)
        #pragma unroll
        for (int r = 0; r < 8; ++r) acc[s][r] = 0.f;

    const int nk = (K + 31) >> 5;
    stageA(0, 0); stageB(0, 0);
    __syncthreads();

    for (int t = 0; t < nk; ++t) {
        const int cb = t & 1;
        if (t + 1 < nk) { stageA(cb ^ 1, (t + 1) << 5); stageB(cb ^ 1, (t + 1) << 5); }

        // A fragment (16x32 bf16): lanes 0-15 -> K {0..7,16..23}, 16-31 -> {8..15,24..31}
        v8u au;
        const int am = (wave << 4) + l16;
        const int akb = half << 3;
        #pragma unroll
        for (int i = 0; i < 4; ++i)
            au[i] = *(const unsigned int*)&As[cb][am][akb + 2 * i];
        #pragma unroll
        for (int i = 0; i < 4; ++i)
            au[4 + i] = *(const unsigned int*)&As[cb][am][akb + 16 + 2 * i];
        BFPack ua; ua.u = au;

        #pragma unroll
        for (int s = 0; s < 4; ++s) {
            v8u bu;
            const int bn = (s << 4) + l16;
            const int bkb = half << 3;
            #pragma unroll
            for (int i = 0; i < 4; ++i) {
                bu[i] = *(const unsigned int*)&Bs[cb][bn][bkb + 2 * i];
                bu[4 + i] = *(const unsigned int*)&Bs[cb][bn][bkb + 16 + 2 * i];
            }
            BFPack ub; ub.u = bu;
            acc[s] = __builtin_amdgcn_wmma_f32_16x16x32_bf16(
                false, ua.b, false, ub.b, (short)0, acc[s], false, false);
        }
        __syncthreads();
    }

    float g = 1.f;
    if (resid) g = 1.f / (1.f + __expf(-gate[0]));
    #pragma unroll
    for (int s = 0; s < 4; ++s) {
        int n = n0 + (s << 4) + l16;
        if (n >= N) continue;
        float bv = bias ? bias[n] : 0.f;
        #pragma unroll
        for (int r = 0; r < 8; ++r) {
            int m = m0 + (wave << 4) + r + (half << 3);
            if (m >= M) continue;
            float v = acc[s][r] + bv;
            if (resid) v = g * v + (1.f - g) * resid[(size_t)m * EMB + n];
            if (act == 1) v = elu1(v);
            C[(size_t)m * N + n] = v;
        }
    }
}

// ---------------------------------------------------------------------------
// Per-head GEMM:  O[:, h*32:(h+1)*32] = X[:, h*32:(h+1)*32] @ R[h]   (R: [H,32,32])
// K=32 -> exactly one 16x16x32 WMMA per 16x16 output subtile.
// Block: 256 thr (8 waves) -> 128 rows; grid = (rows/128, H). rows % 128 == 0.
// ---------------------------------------------------------------------------
__global__ __launch_bounds__(256) void k_head_gemm(
    const float* __restrict__ X, const float* __restrict__ R,
    float* __restrict__ O, int Mrows)
{
    __shared__ __align__(16) unsigned short Xs[128][36];
    __shared__ __align__(16) unsigned short Rs[32][36];   // Rs[n][k] = R[h][k][n]
    const int h = blockIdx.y;
    const int tid = threadIdx.x;
    const int wave = tid >> 5, lane = tid & 31;
    const int half = lane >> 4, l16 = lane & 15;
    const int m0 = blockIdx.x * 128;
    (void)Mrows;

    for (int i = tid; i < 128; i += 256) {        // 8 ngroups x 16 kpairs
        int ng = i & 7, kp = i >> 3;
        int n = ng << 2, k2 = kp << 1;
        const float* rp = R + (size_t)h * 1024 + (size_t)k2 * 32 + n;
        float4 r0 = *(const float4*)rp;
        float4 r1 = *(const float4*)(rp + 32);
        *(unsigned int*)&Rs[n + 0][k2] = pk2bf(r0.x, r1.x);
        *(unsigned int*)&Rs[n + 1][k2] = pk2bf(r0.y, r1.y);
        *(unsigned int*)&Rs[n + 2][k2] = pk2bf(r0.z, r1.z);
        *(unsigned int*)&Rs[n + 3][k2] = pk2bf(r0.w, r1.w);
    }
    for (int i = tid; i < 128 * 8; i += 256) {
        int r = i >> 3, c4 = (i & 7) << 2;
        const float* xp = X + (size_t)(m0 + r) * EMB + h * 32 + c4;
        float4 v = *(const float4*)xp;
        uint2 p; p.x = pk2bf(v.x, v.y); p.y = pk2bf(v.z, v.w);
        *(uint2*)&Xs[r][c4] = p;
    }
    __syncthreads();

    v8f acc[2];
    #pragma unroll
    for (int s = 0; s < 2; ++s)
        #pragma unroll
        for (int r = 0; r < 8; ++r) acc[s][r] = 0.f;

    v8u au;
    const int am = (wave << 4) + l16;
    const int akb = half << 3;
    #pragma unroll
    for (int i = 0; i < 4; ++i) au[i]     = *(const unsigned int*)&Xs[am][akb + 2 * i];
    #pragma unroll
    for (int i = 0; i < 4; ++i) au[4 + i] = *(const unsigned int*)&Xs[am][akb + 16 + 2 * i];
    BFPack ua; ua.u = au;

    #pragma unroll
    for (int s = 0; s < 2; ++s) {
        v8u bu;
        const int bn = (s << 4) + l16;
        const int bkb = half << 3;
        #pragma unroll
        for (int i = 0; i < 4; ++i) {
            bu[i] = *(const unsigned int*)&Rs[bn][bkb + 2 * i];
            bu[4 + i] = *(const unsigned int*)&Rs[bn][bkb + 16 + 2 * i];
        }
        BFPack ub; ub.u = bu;
        acc[s] = __builtin_amdgcn_wmma_f32_16x16x32_bf16(
            false, ua.b, false, ub.b, (short)0, acc[s], false, false);
    }
    #pragma unroll
    for (int s = 0; s < 2; ++s) {
        int n = h * 32 + (s << 4) + l16;
        #pragma unroll
        for (int r = 0; r < 8; ++r) {
            int m = m0 + (wave << 4) + r + (half << 3);
            O[(size_t)m * EMB + n] = acc[s][r];
        }
    }
}

// ---------------------------------------------------------------------------
// Edge attention logits: a[e,h] = dot32(Q[dst], Ka[src]) * p_rel[h] / sqrt(32)
// ---------------------------------------------------------------------------
__global__ void k_edge_attn(const float* __restrict__ Qd, const float* __restrict__ Ka,
                            const float* __restrict__ prel,
                            const int* __restrict__ src, const int* __restrict__ dst,
                            int E, float* __restrict__ aout, int* __restrict__ segout,
                            int offd)
{
    int t = blockIdx.x * blockDim.x + threadIdx.x;
    if (t >= E * HH) return;
    int e = t >> 3, h = t & 7;
    int s = src[e], d = dst[e];
    const float* qp = Qd + (size_t)d * EMB + h * 32;
    const float* kp = Ka + (size_t)s * EMB + h * 32;
    float acc = 0.f;
    #pragma unroll
    for (int i = 0; i < 8; ++i) {
        float4 q4 = *(const float4*)(qp + 4 * i);
        float4 k4 = *(const float4*)(kp + 4 * i);
        acc += q4.x * k4.x + q4.y * k4.y + q4.z * k4.z + q4.w * k4.w;
    }
    aout[(size_t)e * HH + h] = acc * prel[h] * 0.176776695296636881f;
    if (h == 0) segout[e] = offd + d;
}

// fill helpers ---------------------------------------------------------------
__global__ void k_fill_f(float* p, float v, int n) {
    int t = blockIdx.x * 256 + threadIdx.x; if (t < n) p[t] = v;
}
__global__ void k_fill_u(unsigned int* p, unsigned int v, int n) {
    int t = blockIdx.x * 256 + threadIdx.x; if (t < n) p[t] = v;
}

// segment max via order-preserving uint encoding ----------------------------
__global__ void k_edge_max(const float* __restrict__ a, const int* __restrict__ seg,
                           unsigned int* __restrict__ menc, int n)
{
    int t = blockIdx.x * 256 + threadIdx.x;
    if (t >= n) return;
    int e = t >> 3, h = t & 7;
    unsigned int iu = __float_as_uint(a[t]);
    unsigned int u = (iu & 0x80000000u) ? ~iu : (iu | 0x80000000u);
    atomicMax(&menc[(size_t)seg[e] * HH + h], u);
}

__global__ void k_edge_exp(float* __restrict__ a, const int* __restrict__ seg,
                           const unsigned int* __restrict__ menc,
                           float* __restrict__ z, int n)
{
    int t = blockIdx.x * 256 + threadIdx.x;
    if (t >= n) return;
    int e = t >> 3, h = t & 7;
    int sg = seg[e];
    unsigned int u = menc[(size_t)sg * HH + h];
    unsigned int iu = (u & 0x80000000u) ? (u & 0x7FFFFFFFu) : ~u;
    float m = __uint_as_float(iu);
    if ((iu & 0x7F800000u) == 0x7F800000u) m = 0.f;   // empty segment -> 0
    float ev = __expf(a[t] - m);
    a[t] = ev;
    atomicAdd(&z[(size_t)sg * HH + h], ev);
}

__global__ void k_edge_agg(const float* __restrict__ aE, const int* __restrict__ segE,
                           const float* __restrict__ z, const float* __restrict__ Vm,
                           const int* __restrict__ src, float* __restrict__ agg, int E)
{
    int t = blockIdx.x * 256 + threadIdx.x;
    if (t >= E * HH) return;
    int e = t >> 3, h = t & 7;
    int sg = segE[e], s = src[e];
    float alpha = aE[(size_t)e * HH + h] / (z[(size_t)sg * HH + h] + 1e-16f);
    const float* vp = Vm + (size_t)s * EMB + h * 32;
    float* op = agg + (size_t)sg * EMB + h * 32;
    #pragma unroll
    for (int i = 0; i < 8; ++i) {
        float4 v4 = *(const float4*)(vp + 4 * i);
        atomicAdd(&op[4 * i + 0], v4.x * alpha);
        atomicAdd(&op[4 * i + 1], v4.y * alpha);
        atomicAdd(&op[4 * i + 2], v4.z * alpha);
        atomicAdd(&op[4 * i + 3], v4.w * alpha);
    }
}

__global__ void k_gelu(float* __restrict__ x, int n) {
    int t = blockIdx.x * 256 + threadIdx.x;
    if (t >= n) return;
    float v = x[t];
    x[t] = 0.5f * v * (1.f + erff(v * 0.70710678118654752f));
}

// pooling: h_pool[b] = [mean(hc), mean(hp)] ---------------------------------
__global__ __launch_bounds__(256) void k_pool(const float* __restrict__ X,
                                              float* __restrict__ hpool)
{
    int b = blockIdx.x, col = threadIdx.x;
    float s = 0.f;
    for (int i = 0; i < NCN; ++i) s += X[(size_t)(b * NCN + i) * EMB + col];
    hpool[(size_t)b * 512 + col] = s * (1.f / NCN);
    s = 0.f;
    for (int i = 0; i < NPN; ++i) s += X[(size_t)(NCT + b * NPN + i) * EMB + col];
    hpool[(size_t)b * 512 + EMB + col] = s * (1.f / NPN);
}

// ---------------------------------------------------------------------------
// Fused actor tail: h1 = elu(Yc[b,c] + Yp[b,p]) (b1 folded into Yc);
// logits = elu(h1 @ W2 + b2) @ w3 + b3, never materializing h2.
// 64 rows/block, 128 threads (4 waves). Double-buffered W2 staging.
// ---------------------------------------------------------------------------
__global__ __launch_bounds__(128) void k_actor_tail(
    const float* __restrict__ Yc, const float* __restrict__ Yp,
    const float* __restrict__ W2, const float* __restrict__ b2,
    const float* __restrict__ w3, const float* __restrict__ b3,
    float* __restrict__ logits)
{
    __shared__ __align__(16) unsigned short H1s[64][520];   // 64 x 512 bf16
    __shared__ __align__(16) unsigned short Ws[2][64][36];  // Ws[n][k]
    __shared__ float Lacc[64];
    const int tid = threadIdx.x;
    const int wave = tid >> 5, lane = tid & 31;
    const int half = lane >> 4, l16 = lane & 15;
    const long long row0 = (long long)blockIdx.x * 64;

    if (tid < 64) Lacc[tid] = 0.f;

    // Build h1 tile: float4 loads, bf16 pack, ds_store_b64.
    for (int i = tid; i < 64 * 128; i += 128) {
        int r = i >> 7, n4 = (i & 127) << 2;
        long long g = row0 + r;
        int b   = (int)(g >> 11);             // / (NP*NC)=2048
        int rem = (int)(g & 2047);
        int p = rem >> 3, c = rem & 7;
        float4 a4 = *(const float4*)(Yc + (size_t)(b * NCN + c) * 512 + n4);
        float4 b4 = *(const float4*)(Yp + (size_t)(b * NPN + p) * 512 + n4);
        uint2 pk;
        pk.x = pk2bf(elu1(a4.x + b4.x), elu1(a4.y + b4.y));
        pk.y = pk2bf(elu1(a4.z + b4.z), elu1(a4.w + b4.w));
        *(uint2*)&H1s[r][n4] = pk;
    }

    float pl[8];
    #pragma unroll
    for (int r = 0; r < 8; ++r) pl[r] = 0.f;

    for (int nch = 0; nch < 8; ++nch) {
        auto stageW = [&](int bf_, int kk) {
            // 16 ngroups x 16 kpairs = 256 units, 128 threads -> 2 each
            for (int i = tid; i < 256; i += 128) {
                int ng = i & 15, kp = i >> 4;
                int n = ng << 2, k2 = kp << 1;
                const float* w0 = W2 + (size_t)(kk + k2) * 512 + nch * 64 + n;
                float4 r0 = *(const float4*)w0;
                float4 r1 = *(const float4*)(w0 + 512);
                *(unsigned int*)&Ws[bf_][n + 0][k2] = pk2bf(r0.x, r1.x);
                *(unsigned int*)&Ws[bf_][n + 1][k2] = pk2bf(r0.y, r1.y);
                *(unsigned int*)&Ws[bf_][n + 2][k2] = pk2bf(r0.z, r1.z);
                *(unsigned int*)&Ws[bf_][n + 3][k2] = pk2bf(r0.w, r1.w);
            }
        };

        v8f acc[4];
        #pragma unroll
        for (int s = 0; s < 4; ++s)
            #pragma unroll
            for (int r = 0; r < 8; ++r) acc[s][r] = 0.f;

        stageW(0, 0);
        __syncthreads();                       // also covers H1s/Lacc on nch==0
        for (int t = 0; t < 16; ++t) {
            const int cb = t & 1;
            if (t + 1 < 16) stageW(cb ^ 1, (t + 1) << 5);

            v8u au;
            const int am = (wave << 4) + l16;
            const int akb = (t << 5) + (half << 3);
            #pragma unroll
            for (int i = 0; i < 4; ++i) au[i]     = *(const unsigned int*)&H1s[am][akb + 2 * i];
            #pragma unroll
            for (int i = 0; i < 4; ++i) au[4 + i] = *(const unsigned int*)&H1s[am][akb + 16 + 2 * i];
            BFPack ua; ua.u = au;

            #pragma unroll
            for (int s = 0; s < 4; ++s) {
                v8u bu;
                const int bn = (s << 4) + l16;
                const int bkb = half << 3;
                #pragma unroll
                for (int i = 0; i < 4; ++i) {
                    bu[i] = *(const unsigned int*)&Ws[cb][bn][bkb + 2 * i];
                    bu[4 + i] = *(const unsigned int*)&Ws[cb][bn][bkb + 16 + 2 * i];
                }
                BFPack ub; ub.u = bu;
                acc[s] = __builtin_amdgcn_wmma_f32_16x16x32_bf16(
                    false, ua.b, false, ub.b, (short)0, acc[s], false, false);
            }
            __syncthreads();
        }
        #pragma unroll
        for (int s = 0; s < 4; ++s) {
            int n = nch * 64 + (s << 4) + l16;
            float bv = b2[n], wv = w3[n];
            #pragma unroll
            for (int r = 0; r < 8; ++r)
                pl[r] += elu1(acc[s][r] + bv) * wv;
        }
    }
    #pragma unroll
    for (int r = 0; r < 8; ++r)
        atomicAdd(&Lacc[(wave << 4) + r + (half << 3)], pl[r]);
    __syncthreads();
    if (tid < 64) logits[row0 + tid] = Lacc[tid] + b3[0];
}

// ---------------------------------------------------------------------------
// Masked log-softmax per batch row (2048 logits), action log-prob + entropy.
// mask layout: [B, NC, NP]; logit index a = p*NC + c.
// ---------------------------------------------------------------------------
__global__ __launch_bounds__(256) void k_logsoftmax(
    const float* __restrict__ logits, const int* __restrict__ mask,
    const int* __restrict__ action, float* __restrict__ alp, float* __restrict__ ent)
{
    const int b = blockIdx.x, tid = threadIdx.x;
    __shared__ float red[256];
    const float* lg = logits + (size_t)b * 2048;
    const int* mk = mask + (size_t)b * 2048;

    float mx = -3.4e38f;
    for (int a = tid; a < 2048; a += 256) {
        int p = a >> 3, c = a & 7;
        if (mk[c * NPN + p]) mx = fmaxf(mx, lg[a]);
    }
    red[tid] = mx; __syncthreads();
    for (int s = 128; s > 0; s >>= 1) { if (tid < s) red[tid] = fmaxf(red[tid], red[tid + s]); __syncthreads(); }
    mx = red[0]; __syncthreads();

    float se = 0.f;
    for (int a = tid; a < 2048; a += 256) {
        int p = a >> 3, c = a & 7;
        if (mk[c * NPN + p]) se += __expf(lg[a] - mx);
    }
    red[tid] = se; __syncthreads();
    for (int s = 128; s > 0; s >>= 1) { if (tid < s) red[tid] += red[tid + s]; __syncthreads(); }
    float lz = logf(red[0]); __syncthreads();

    float el = 0.f;
    for (int a = tid; a < 2048; a += 256) {
        int p = a >> 3, c = a & 7;
        if (mk[c * NPN + p]) {
            float lp = lg[a] - mx - lz;
            float pr = __expf(lp);
            if (pr > 0.f) el -= pr * lp;
        }
    }
    red[tid] = el; __syncthreads();
    for (int s = 128; s > 0; s >>= 1) { if (tid < s) red[tid] += red[tid + s]; __syncthreads(); }
    if (tid == 0) {
        ent[b] = red[0];
        alp[b] = lg[action[b]] - mx - lz;
    }
}

// critic head: out[b] = dot(X[b,:], w) + bias -------------------------------
__global__ __launch_bounds__(128) void k_rowdot(const float* __restrict__ X,
                                                const float* __restrict__ w,
                                                const float* __restrict__ bias,
                                                float* __restrict__ out, int Kd)
{
    const int b = blockIdx.x, tid = threadIdx.x;
    __shared__ float red[128];
    const float* x = X + (size_t)b * Kd;
    float s = 0.f;
    for (int i = tid; i < Kd; i += 128) s += x[i] * w[i];
    red[tid] = s; __syncthreads();
    for (int st = 64; st > 0; st >>= 1) { if (tid < st) red[tid] += red[tid + st]; __syncthreads(); }
    if (tid == 0) out[b] = red[0] + bias[0];
}

// ---------------------------------------------------------------------------
// Host orchestration
// ---------------------------------------------------------------------------
struct LayerP {
    const float *alw_c, *alb_c, *alw_p, *alb_p;
    const float *arel[3], *mrel[3], *prel[3];
    const float *kw_c, *kb_c, *kw_p, *kb_p;
    const float *qw_c, *qb_c, *qw_p, *qb_p;
    const float *vw_c, *vb_c, *vw_p, *vb_p;
    const float *skip_c, *skip_p;
};

static inline int gdiv(int a, int b) { return (a + b - 1) / b; }

extern "C" void kernel_launch(void* const* d_in, const int* in_sizes, int n_in,
                              void* d_out, int out_size, void* d_ws, size_t ws_size,
                              hipStream_t stream)
{
    (void)in_sizes; (void)n_in; (void)out_size; (void)ws_size;

    const float* x_crane = (const float*)d_in[0];
    const float* x_pile  = (const float*)d_in[1];
    const int* ec2p = (const int*)d_in[2];
    const int* ep2c = (const int*)d_in[3];
    const int* ep2p = (const int*)d_in[4];
    const int* action = (const int*)d_in[5];
    const int* mask   = (const int*)d_in[6];

    // Leaves flattened per jax pytree order (dict keys sorted within each node).
    int cur = 7;
    auto nxt = [&]() { return (const float*)d_in[cur++]; };
    LayerP L[2];
    for (int l = 0; l < 2; ++l) {
        L[l].alw_c = nxt(); L[l].alb_c = nxt(); L[l].alw_p = nxt(); L[l].alb_p = nxt();
        for (int i = 0; i < 3; ++i) L[l].arel[i] = nxt();     // c2p, p2c, p2p
        L[l].kw_c = nxt(); L[l].kb_c = nxt(); L[l].kw_p = nxt(); L[l].kb_p = nxt();
        for (int i = 0; i < 3; ++i) L[l].mrel[i] = nxt();
        for (int i = 0; i < 3; ++i) L[l].prel[i] = nxt();
        L[l].qw_c = nxt(); L[l].qb_c = nxt(); L[l].qw_p = nxt(); L[l].qb_p = nxt();
        L[l].skip_c = nxt(); L[l].skip_p = nxt();
        L[l].vw_c = nxt(); L[l].vb_c = nxt(); L[l].vw_p = nxt(); L[l].vb_p = nxt();
    }
    const float *aw1 = nxt(), *ab1 = nxt(), *aw2 = nxt(), *ab2 = nxt(), *aw3 = nxt(), *ab3 = nxt();
    const float *cw1 = nxt(), *cb1 = nxt(), *cw2 = nxt(), *cb2 = nxt(), *cw3 = nxt(), *cb3 = nxt();

    // ---- workspace carve ----
    char* wp = (char*)d_ws;
    auto carve = [&](size_t bytes) { void* p = wp; wp += (bytes + 255) & ~(size_t)255; return p; };
    float* XA   = (float*)carve((size_t)TOT * EMB * 4);
    float* XB   = (float*)carve((size_t)TOT * EMB * 4);
    float* Kb   = (float*)carve((size_t)TOT * EMB * 4);
    float* Qb   = (float*)carve((size_t)TOT * EMB * 4);
    float* Vb   = (float*)carve((size_t)TOT * EMB * 4);
    float* Ka0  = (float*)carve((size_t)NCT * EMB * 4);
    float* Ka1  = (float*)carve((size_t)NPT * EMB * 4);
    float* Ka2  = (float*)carve((size_t)NPT * EMB * 4);
    float* Vm0  = (float*)carve((size_t)NCT * EMB * 4);
    float* Vm1  = (float*)carve((size_t)NPT * EMB * 4);
    float* Vm2  = (float*)carve((size_t)NPT * EMB * 4);
    float* abuf = (float*)carve((size_t)3 * EPT * HH * 4);
    int*   segb = (int*)  carve((size_t)3 * EPT * 4);
    unsigned int* menc = (unsigned int*)carve((size_t)TOT * HH * 4);
    float* zb   = (float*)carve((size_t)TOT * HH * 4);
    float* agg  = (float*)carve((size_t)TOT * EMB * 4);
    float* Yc   = (float*)carve((size_t)NCT * 512 * 4);
    float* Yp   = (float*)carve((size_t)NPT * 512 * 4);
    float* logits = (float*)carve((size_t)NROWS * 4);
    float* hpool  = (float*)carve((size_t)BB * 512 * 4);
    float* cr1    = (float*)carve((size_t)BB * 512 * 4);
    float* cr2    = (float*)carve((size_t)BB * 512 * 4);

    float* alp_out = (float*)d_out;
    float* val_out = alp_out + BB;
    float* ent_out = alp_out + 2 * BB;

    auto gemm = [&](const float* A, const float* W, const float* bias,
                    const float* resid, const float* gate, float* C,
                    int M, int K, int N, int act) {
        dim3 g(gdiv(M, 128), gdiv(N, 64));
        k_gemm<<<g, 256, 0, stream>>>(A, W, bias, resid, gate, C, M, K, N, act);
    };
    auto head = [&](const float* X, const float* R, float* O, int rows) {
        dim3 g(rows / 128, HH);
        k_head_gemm<<<g, 256, 0, stream>>>(X, R, O, rows);
    };

    const float* xin_c = x_crane; int KC = 40;
    const float* xin_p = x_pile;  int KP = 40;
    float* outs[2] = { XA, XB };

    for (int l = 0; l < 2; ++l) {
        const LayerP& P = L[l];
        float* Xout = outs[l];

        // K/Q/V projections (crane rows [0,NCT), pile rows [NCT,TOT))
        gemm(xin_c, P.kw_c, P.kb_c, nullptr, nullptr, Kb,               NCT, KC, EMB, 0);
        gemm(xin_c, P.qw_c, P.qb_c, nullptr, nullptr, Qb,               NCT, KC, EMB, 0);
        gemm(xin_c, P.vw_c, P.vb_c, nullptr, nullptr, Vb,               NCT, KC, EMB, 0);
        gemm(xin_p, P.kw_p, P.kb_p, nullptr, nullptr, Kb + (size_t)NCT * EMB, NPT, KP, EMB, 0);
        gemm(xin_p, P.qw_p, P.qb_p, nullptr, nullptr, Qb + (size_t)NCT * EMB, NPT, KP, EMB, 0);
        gemm(xin_p, P.vw_p, P.vb_p, nullptr, nullptr, Vb + (size_t)NCT * EMB, NPT, KP, EMB, 0);

        // Hoisted per-node relation transforms (ke/ve depend only on src node)
        head(Kb,                      P.arel[0], Ka0, NCT);   // c2p: src crane
        head(Vb,                      P.mrel[0], Vm0, NCT);
        head(Kb + (size_t)NCT * EMB,  P.arel[1], Ka1, NPT);   // p2c: src pile
        head(Vb + (size_t)NCT * EMB,  P.mrel[1], Vm1, NPT);
        head(Kb + (size_t)NCT * EMB,  P.arel[2], Ka2, NPT);   // p2p: src pile
        head(Vb + (size_t)NCT * EMB,  P.mrel[2], Vm2, NPT);

        // Edge attention logits + global segment ids
        int eb = gdiv(EPT * HH, 256);
        k_edge_attn<<<eb, 256, 0, stream>>>(Qb + (size_t)NCT * EMB, Ka0, P.prel[0],
                                            ec2p, ec2p + EPT, EPT, abuf, segb, NCT);
        k_edge_attn<<<eb, 256, 0, stream>>>(Qb, Ka1, P.prel[1],
                                            ep2c, ep2c + EPT, EPT,
                                            abuf + (size_t)EPT * HH, segb + EPT, 0);
        k_edge_attn<<<eb, 256, 0, stream>>>(Qb + (size_t)NCT * EMB, Ka2, P.prel[2],
                                            ep2p, ep2p + EPT, EPT,
                                            abuf + (size_t)2 * EPT * HH, segb + 2 * EPT, NCT);

        // Segment softmax + aggregation
        k_fill_u<<<gdiv(TOT * HH, 256), 256, 0, stream>>>(menc, 0u, TOT * HH);
        k_fill_f<<<gdiv(TOT * HH, 256), 256, 0, stream>>>(zb, 0.f, TOT * HH);
        k_fill_f<<<gdiv(TOT * EMB, 256), 256, 0, stream>>>(agg, 0.f, TOT * EMB);

        int nall = 3 * EPT * HH;
        k_edge_max<<<gdiv(nall, 256), 256, 0, stream>>>(abuf, segb, menc, nall);
        k_edge_exp<<<gdiv(nall, 256), 256, 0, stream>>>(abuf, segb, menc, zb, nall);
        k_edge_agg<<<eb, 256, 0, stream>>>(abuf, segb, zb, Vm0, ec2p, agg, EPT);
        k_edge_agg<<<eb, 256, 0, stream>>>(abuf + (size_t)EPT * HH, segb + EPT, zb, Vm1, ep2c, agg, EPT);
        k_edge_agg<<<eb, 256, 0, stream>>>(abuf + (size_t)2 * EPT * HH, segb + 2 * EPT, zb, Vm2, ep2p, agg, EPT);

        // gelu -> a_lin (+ gated skip on layer 1) -> elu
        k_gelu<<<gdiv(TOT * EMB, 256), 256, 0, stream>>>(agg, TOT * EMB);
        const float* rc = (l == 0) ? nullptr : xin_c;
        const float* rp = (l == 0) ? nullptr : xin_p;
        gemm(agg,                      P.alw_c, P.alb_c, rc, P.skip_c, Xout,                      NCT, EMB, EMB, 1);
        gemm(agg + (size_t)NCT * EMB,  P.alw_p, P.alb_p, rp, P.skip_p, Xout + (size_t)NCT * EMB,  NPT, EMB, EMB, 1);

        xin_c = Xout; xin_p = Xout + (size_t)NCT * EMB; KC = KP = EMB;
    }

    // Pooling
    k_pool<<<BB, 256, 0, stream>>>(XB, hpool);

    // Actor layer 1, factorized: Yc = hc@W1[:256]+b1, Yp = hp@W1[256:]
    gemm(XB,                      aw1,               ab1,     nullptr, nullptr, Yc, NCT, EMB, 512, 0);
    gemm(XB + (size_t)NCT * EMB,  aw1 + 256 * 512,   nullptr, nullptr, nullptr, Yp, NPT, EMB, 512, 0);

    // Fused actor layer 2 + logit head
    k_actor_tail<<<NROWS / 64, 128, 0, stream>>>(Yc, Yp, aw2, ab2, aw3, ab3, logits);

    // Masked log-softmax -> alp, ent
    k_logsoftmax<<<BB, 256, 0, stream>>>(logits, mask, action, alp_out, ent_out);

    // Critic
    gemm(hpool, cw1, cb1, nullptr, nullptr, cr1, BB, 512, 512, 1);
    gemm(cr1,   cw2, cb2, nullptr, nullptr, cr2, BB, 512, 512, 1);
    k_rowdot<<<BB, 128, 0, stream>>>(cr2, cw3, cb3, val_out, 512);
}
